// DocumentLevelSelfAttention_82738249990374
// MI455X (gfx1250) — compile-verified, matrix operand order checked by reading, not run on hardware
//
#include <hip/hip_runtime.h>

#define B_   32
#define N_   2048
#define TWOD 1024
#define FTOT 1536
#define CAT_ 512
#define DA   350
#define DAP  352
#define R_   4

typedef __attribute__((ext_vector_type(16))) __bf16 bf16x16;
typedef __attribute__((ext_vector_type(8)))  float  f32x8;

union FragA { bf16x16 v; unsigned short e[16]; };
union FragB { bf16x16 v; uint4 q[2]; };

__device__ __forceinline__ unsigned short f2bf_u16(float f) {
  unsigned int u = __builtin_bit_cast(unsigned int, f);
  u += 0x7FFFu + ((u >> 16) & 1u);           // round-to-nearest-even
  return (unsigned short)(u >> 16);
}

// ---------------------------------------------------------------- kernel 1:
// S1_w[:, :1024] -> bf16, padded to 352 rows (rows 350,351 = 0)
__global__ void k_convert_s1(const float* __restrict__ S1w,
                             unsigned short* __restrict__ s1h) {
  int idx = blockIdx.x * 256 + threadIdx.x;   // 352*1024 elements
  int d = idx >> 10;
  int k = idx & 1023;
  float v = (d < DA) ? S1w[d * FTOT + k] : 0.0f;
  s1h[idx] = f2bf_u16(v);
}

// ---------------------------------------------------------------- kernel 2:
// bias[b][d] = sum_c cat[b][c] * S1_w[d][1024+c]   (padded to 352)
__global__ void k_bias(const float* __restrict__ cat,
                       const float* __restrict__ S1w,
                       float* __restrict__ biasws) {
  int b = blockIdx.x;
  for (int d = threadIdx.x; d < DAP; d += 256) {
    float acc = 0.0f;
    if (d < DA) {
      const float* wrow = S1w + d * FTOT + TWOD;
      const float* crow = cat + b * CAT_;
      for (int c = 0; c < CAT_; ++c) acc = fmaf(crow[c], wrow[c], acc);
    }
    biasws[b * DAP + d] = acc;
  }
}

// ---------------------------------------------------------------- kernel 3:
// scores[b][n][r] = S2( tanh( H[b,n,:] @ S1_H^T + bias[b] ) )
// block = (b, 64-row tile). 8 waves = 4 row-groups x 2 col-groups(11 d-tiles).
__global__ __launch_bounds__(256)
void k_scores(const float* __restrict__ H,
              const unsigned short* __restrict__ s1h,
              const float* __restrict__ biasws,
              const float* __restrict__ S2w,
              float* __restrict__ scores) {
  const int b    = blockIdx.x;
  const int nblk = blockIdx.y;          // 32 tiles of 64 rows
  const int tid  = threadIdx.x;
  const int wave = tid >> 5;
  const int lane = tid & 31;
  const int rg   = wave >> 1;           // row-group 0..3
  const int cg   = wave & 1;            // col-group 0..1
  const int row16 = lane & 15;
  const int koff  = (lane >> 4) << 3;   // 0 or 8 (wave32 A/B K-layout)

  const int n_row = nblk * 64 + rg * 16 + row16;
  const float* hrow = H + ((size_t)b * N_ + n_row) * TWOD;
  const unsigned short* wbase = s1h + (size_t)(cg * 176 + row16) * TWOD;

  f32x8 acc[11];
  f32x8 zero = {0.f,0.f,0.f,0.f,0.f,0.f,0.f,0.f};
  #pragma unroll
  for (int j = 0; j < 11; ++j) acc[j] = zero;

  #pragma unroll 1
  for (int ks = 0; ks < TWOD; ks += 32) {
    // A fragment: 16x32 bf16 tile of H rows, converted on the fly
    FragA a;
    {
      const float* p0 = hrow + ks + koff;
      float4 f0 = *(const float4*)(p0);
      float4 f1 = *(const float4*)(p0 + 4);
      float4 f2 = *(const float4*)(p0 + 16);
      float4 f3 = *(const float4*)(p0 + 20);
      a.e[ 0]=f2bf_u16(f0.x); a.e[ 1]=f2bf_u16(f0.y); a.e[ 2]=f2bf_u16(f0.z); a.e[ 3]=f2bf_u16(f0.w);
      a.e[ 4]=f2bf_u16(f1.x); a.e[ 5]=f2bf_u16(f1.y); a.e[ 6]=f2bf_u16(f1.z); a.e[ 7]=f2bf_u16(f1.w);
      a.e[ 8]=f2bf_u16(f2.x); a.e[ 9]=f2bf_u16(f2.y); a.e[10]=f2bf_u16(f2.z); a.e[11]=f2bf_u16(f2.w);
      a.e[12]=f2bf_u16(f3.x); a.e[13]=f2bf_u16(f3.y); a.e[14]=f2bf_u16(f3.z); a.e[15]=f2bf_u16(f3.w);
    }
    #pragma unroll
    for (int jj = 0; jj < 11; ++jj) {
      FragB bf;
      const unsigned short* wp = wbase + (size_t)(jj * 16) * TWOD + ks + koff;
      bf.q[0] = *(const uint4*)(wp);        // K  .. K+7
      bf.q[1] = *(const uint4*)(wp + 16);   // K+16 .. K+23
      acc[jj] = __builtin_amdgcn_wmma_f32_16x16x32_bf16(
          false, a.v, false, bf.v, (short)0, acc[jj], false, false);
    }
  }

  // Epilogue: +bias, tanh, contract with S2 (per-lane partials over d).
  // MUST be fully unrolled so acc[] stays in registers (no scratch spill).
  float ps[8][4];
  #pragma unroll
  for (int i = 0; i < 8; ++i)
    #pragma unroll
    for (int r = 0; r < 4; ++r) ps[i][r] = 0.f;

  #pragma unroll
  for (int jj = 0; jj < 11; ++jj) {
    int d = (cg * 11 + jj) * 16 + row16;
    float bias = biasws[b * DAP + d];
    float w0 = (d < DA) ? S2w[0 * DA + d] : 0.f;
    float w1 = (d < DA) ? S2w[1 * DA + d] : 0.f;
    float w2 = (d < DA) ? S2w[2 * DA + d] : 0.f;
    float w3 = (d < DA) ? S2w[3 * DA + d] : 0.f;
    #pragma unroll
    for (int i = 0; i < 8; ++i) {
      float t = tanhf(acc[jj][i] + bias);
      ps[i][0] = fmaf(t, w0, ps[i][0]);
      ps[i][1] = fmaf(t, w1, ps[i][1]);
      ps[i][2] = fmaf(t, w2, ps[i][2]);
      ps[i][3] = fmaf(t, w3, ps[i][3]);
    }
  }
  // reduce the d-sum across each 16-lane half (wave32 C layout)
  #pragma unroll
  for (int i = 0; i < 8; ++i)
    #pragma unroll
    for (int r = 0; r < 4; ++r) {
      float v = ps[i][r];
      v += __shfl_xor(v, 1, 32);
      v += __shfl_xor(v, 2, 32);
      v += __shfl_xor(v, 4, 32);
      v += __shfl_xor(v, 8, 32);
      ps[i][r] = v;
    }

  __shared__ float s2lds[4][2][16][4];
  if ((lane & 15) == 0) {
    int half = lane >> 4;  // lanes 0..15 -> rows 0..7, lanes 16..31 -> rows 8..15
    #pragma unroll
    for (int i = 0; i < 8; ++i)
      #pragma unroll
      for (int r = 0; r < 4; ++r)
        s2lds[rg][cg][half * 8 + i][r] = ps[i][r];
  }
  __syncthreads();
  {
    int rg2 = tid >> 6, row = (tid >> 2) & 15, r = tid & 3;
    float val = s2lds[rg2][0][row][r] + s2lds[rg2][1][row][r];
    int n = nblk * 64 + rg2 * 16 + row;
    scores[((size_t)b * N_ + n) * R_ + r] = val;
  }
}

// ---------------------------------------------------------------- kernel 4:
// masked softmax over n for each (b,r); writes weights output [b][r][n]
__global__ void k_softmax(const float* __restrict__ scores,
                          const int* __restrict__ lengths,
                          float* __restrict__ weights) {
  int b = blockIdx.x;
  int tid = threadIdx.x;
  int lane = tid & 31, wid = tid >> 5;
  int len = lengths[b];

  float4 vals[8];
  #pragma unroll
  for (int i = 0; i < 8; ++i) {
    int n = tid + i * 256;
    vals[i] = *(const float4*)(scores + ((size_t)b * N_ + n) * R_);
  }

  float mx[4] = {-1e30f, -1e30f, -1e30f, -1e30f};
  #pragma unroll
  for (int i = 0; i < 8; ++i) {
    int n = tid + i * 256;
    if (n < len) {
      mx[0] = fmaxf(mx[0], vals[i].x);
      mx[1] = fmaxf(mx[1], vals[i].y);
      mx[2] = fmaxf(mx[2], vals[i].z);
      mx[3] = fmaxf(mx[3], vals[i].w);
    }
  }
  __shared__ float red[8][4];
  #pragma unroll
  for (int r = 0; r < 4; ++r)
    for (int off = 16; off > 0; off >>= 1)
      mx[r] = fmaxf(mx[r], __shfl_xor(mx[r], off, 32));
  if (lane == 0)
    for (int r = 0; r < 4; ++r) red[wid][r] = mx[r];
  __syncthreads();
  #pragma unroll
  for (int r = 0; r < 4; ++r) {
    float m = red[0][r];
    for (int w = 1; w < 8; ++w) m = fmaxf(m, red[w][r]);
    mx[r] = m;
  }
  __syncthreads();

  float sm[4] = {0.f, 0.f, 0.f, 0.f};
  #pragma unroll
  for (int i = 0; i < 8; ++i) {
    int n = tid + i * 256;
    bool ok = n < len;
    float e0 = ok ? __expf(vals[i].x - mx[0]) : 0.f;
    float e1 = ok ? __expf(vals[i].y - mx[1]) : 0.f;
    float e2 = ok ? __expf(vals[i].z - mx[2]) : 0.f;
    float e3 = ok ? __expf(vals[i].w - mx[3]) : 0.f;
    vals[i].x = e0; vals[i].y = e1; vals[i].z = e2; vals[i].w = e3;
    sm[0] += e0; sm[1] += e1; sm[2] += e2; sm[3] += e3;
  }
  #pragma unroll
  for (int r = 0; r < 4; ++r)
    for (int off = 16; off > 0; off >>= 1)
      sm[r] += __shfl_xor(sm[r], off, 32);
  if (lane == 0)
    for (int r = 0; r < 4; ++r) red[wid][r] = sm[r];
  __syncthreads();
  float inv[4];
  #pragma unroll
  for (int r = 0; r < 4; ++r) {
    float s = 0.f;
    for (int w = 0; w < 8; ++w) s += red[w][r];
    inv[r] = 1.0f / s;
  }
  #pragma unroll
  for (int i = 0; i < 8; ++i) {
    int n = tid + i * 256;
    weights[((size_t)b * R_ + 0) * N_ + n] = vals[i].x * inv[0];
    weights[((size_t)b * R_ + 1) * N_ + n] = vals[i].y * inv[1];
    weights[((size_t)b * R_ + 2) * N_ + n] = vals[i].z * inv[2];
    weights[((size_t)b * R_ + 3) * N_ + n] = vals[i].w * inv[3];
  }
}

// ---------------------------------------------------------------- kernel 5:
// partial M: part[b][nc][r][d] = sum_{n in chunk} A[b][r][n] * H[b][n][d]
__global__ void k_pool(const float* __restrict__ H,
                       const float* __restrict__ weights,
                       float* __restrict__ part) {
  int b = blockIdx.x, dc = blockIdx.y, nc = blockIdx.z;
  int tid = threadIdx.x;
  __shared__ float aw[4][512];
  for (int i = tid; i < 2048; i += 256) {
    int r = i >> 9, n = i & 511;
    aw[r][n] = weights[((size_t)b * R_ + r) * N_ + nc * 512 + n];
  }
  __syncthreads();
  int d = dc * 256 + tid;
  const float* hp = H + ((size_t)b * N_ + nc * 512) * TWOD + d;
  float a0 = 0.f, a1 = 0.f, a2 = 0.f, a3 = 0.f;
  for (int n = 0; n < 512; ++n) {
    float h = hp[(size_t)n * TWOD];
    a0 = fmaf(h, aw[0][n], a0);
    a1 = fmaf(h, aw[1][n], a1);
    a2 = fmaf(h, aw[2][n], a2);
    a3 = fmaf(h, aw[3][n], a3);
  }
  size_t base = (((size_t)b * 4 + nc) * R_) * 1024 + d;
  part[base + 0 * 1024] = a0;
  part[base + 1 * 1024] = a1;
  part[base + 2 * 1024] = a2;
  part[base + 3 * 1024] = a3;
}

// ---------------------------------------------------------------- kernel 6:
__global__ void k_bm(const float* __restrict__ part, float* __restrict__ BM) {
  int idx = blockIdx.x * 256 + threadIdx.x;   // 131072 outputs
  int b = idx >> 12;
  int rd = idx & 4095;
  int r = rd >> 10, d = rd & 1023;
  float s = 0.f;
  #pragma unroll
  for (int nc = 0; nc < 4; ++nc)
    s += part[(((size_t)b * 4 + nc) * R_ + r) * 1024 + d];
  BM[idx] = s;
}

extern "C" void kernel_launch(void* const* d_in, const int* in_sizes, int n_in,
                              void* d_out, int out_size, void* d_ws, size_t ws_size,
                              hipStream_t stream) {
  const float* H    = (const float*)d_in[0];
  const float* cat  = (const float*)d_in[1];
  const float* S1w  = (const float*)d_in[2];
  const float* S2w  = (const float*)d_in[3];
  const int*   lens = (const int*)d_in[4];

  char* w = (char*)d_ws;
  unsigned short* s1h = (unsigned short*)w;            // 352*1024*2  = 720896 B
  float* biasws = (float*)(w + 720896);                // 32*352*4    =  45056 B
  float* scores = (float*)(w + 765952);                // 32*2048*4*4 = 1048576 B
  float* part   = (float*)(w + 1814528);               // 32*4*4*1024*4 = 2097152 B

  float* BM      = (float*)d_out;                      // [32, 4096]
  float* weights = BM + (size_t)B_ * R_ * TWOD;        // [32, 4, 2048]

  k_convert_s1<<<1408, 256, 0, stream>>>(S1w, s1h);
  k_bias<<<B_, 256, 0, stream>>>(cat, S1w, biasws);
  k_scores<<<dim3(B_, 32), 256, 0, stream>>>(H, s1h, biasws, S2w, scores);
  k_softmax<<<B_, 256, 0, stream>>>(scores, lens, weights);
  k_pool<<<dim3(B_, 4, 4), 256, 0, stream>>>(H, weights, part);
  k_bm<<<512, 256, 0, stream>>>(part, BM);
}
